// UniversalNeuralNetwork_50964081934811
// MI455X (gfx1250) — matrix operand order repaired and checked by reading, not compiled, not run on hardware
//
#include <hip/hip_runtime.h>
#include <hip/hip_bf16.h>
#include <math.h>

#define BATCH 32768
#define DIM   256
#define DEPTH 9
#define DD    (DIM * DIM)

typedef __attribute__((ext_vector_type(16))) __bf16 v16bf;
typedef __attribute__((ext_vector_type(8)))  __bf16 v8bf;   // 16 bytes -> global_load_b128
typedef __attribute__((ext_vector_type(8)))  float  v8f;

union ABfrag { v16bf v; v8bf h[2]; };

// ---------------------------------------------------------------------------
// Weight prep: f32 [DIM,DIM] -> bf16 Bt layout, Bt[j][k] = weight for output j,
// input k.  transpose=0: src already [j,k] (lin layers, x @ W.T).
// transpose=1: src is [k,j] (universal cw, h @ cw).
// ---------------------------------------------------------------------------
__global__ void prep_w_kernel(const float* __restrict__ src,
                              __bf16* __restrict__ dst, int transpose) {
  int idx = blockIdx.x * blockDim.x + threadIdx.x;   // DD threads
  int j = idx >> 8, k = idx & 255;
  float v = transpose ? src[(k << 8) | j] : src[idx];
  dst[idx] = (__bf16)v;
}

// ---------------------------------------------------------------------------
// Convert input activations f32 -> bf16 (4 elements / thread).
// ---------------------------------------------------------------------------
__global__ void cvt_x_kernel(const float* __restrict__ src,
                             __bf16* __restrict__ dst) {
  int idx = (blockIdx.x * blockDim.x + threadIdx.x) * 4;
  float4 f = *(const float4*)(src + idx);
  dst[idx + 0] = (__bf16)f.x;
  dst[idx + 1] = (__bf16)f.y;
  dst[idx + 2] = (__bf16)f.z;
  dst[idx + 3] = (__bf16)f.w;
}

// ---------------------------------------------------------------------------
// Universal-layer p[j]: 27-step circulant recurrence on a 256-long state.
// One block per output j, one state element per thread, neighbors via LDS.
//   s'[i] = cos(a)*s[i] + sin(a)*(s[i-1] - s[i+1])   (cyclic)
// p[j] = s[0]^2 after all steps.  Independent of batch data.
// ---------------------------------------------------------------------------
__global__ void universal_p_kernel(const float* __restrict__ uw,
                                   float* __restrict__ p) {
  __shared__ float sh[DIM];
  const int j = blockIdx.x;
  const int i = threadIdx.x;
  float s = 0.0625f;                       // 1/sqrt(256)
  sh[i] = s;
  __syncthreads();
  for (int t = 0; t < 3 * DEPTH; ++t) {
    // angles[t, j] = uw[t/3, j, t%3]
    float ang = uw[(t / 3) * DD + j * DIM + (t % 3)];
    float sn, c;
    sincosf(ang, &sn, &c);
    float left  = sh[(i + DIM - 1) & (DIM - 1)];
    float right = sh[(i + 1) & (DIM - 1)];
    __syncthreads();
    s = c * s + sn * (left - right);
    sh[i] = s;
    __syncthreads();
  }
  if (i == 0) p[j] = s * s;
}

// ---------------------------------------------------------------------------
// bf16 WMMA GEMM + fused epilogue.
//   out[m,n] = act( sum_k A[m,k]*Bt[n,k] + bias[n] (+ p[n]) )
// padd != null  -> tanh epilogue (universal layer), else ReLU.
// out_f != null -> write f32 (final layer), else write bf16 activations.
// Block: 256 threads = 8 waves; block tile 128(M) x 64(N); wave tile 16x64.
// ---------------------------------------------------------------------------
__global__ __launch_bounds__(256) void gemm_kernel(
    const __bf16* __restrict__ A, const __bf16* __restrict__ Bt,
    const float* __restrict__ bias, const float* __restrict__ padd,
    __bf16* __restrict__ out_bf, float* __restrict__ out_f) {
  const int lane  = threadIdx.x & 31;
  const int wave  = threadIdx.x >> 5;
  const int lan15 = lane & 15;
  const int hi    = lane >> 4;             // 0 for lanes 0-15, 1 for 16-31

  const int m_base = blockIdx.x * 128 + wave * 16;
  const int n_base = blockIdx.y * 64;

  v8f acc[4] = {v8f{}, v8f{}, v8f{}, v8f{}};

  // A fragment (16x32 bf16): lane row = m_base+lan15; lanes 0-15 take K chunks
  // {k, k+16}, lanes 16-31 take {k+8, k+24}  (ISA 16-bit A layout).
  const __bf16* Arow = A + (size_t)(m_base + lan15) * DIM;
  // B fragment (32x16 bf16): lane col = n+lan15; lanes 0-15 take K chunks
  // {k, k+16}, lanes 16-31 take {k+8, k+24} (matches A).
  const __bf16* Bcol = Bt + (size_t)(n_base + lan15) * DIM + hi * 8;

  for (int k = 0; k < DIM; k += 32) {
    ABfrag a;
    const __bf16* ap = Arow + k + hi * 8;
    a.h[0] = *(const v8bf*)(ap);
    a.h[1] = *(const v8bf*)(ap + 16);
#pragma unroll
    for (int nt = 0; nt < 4; ++nt) {
      ABfrag b;
      const __bf16* bp = Bcol + nt * (16 * DIM) + k;
      b.h[0] = *(const v8bf*)(bp);
      b.h[1] = *(const v8bf*)(bp + 16);
      acc[nt] = __builtin_amdgcn_wmma_f32_16x16x32_bf16(
          false, a.v, false, b.v, (short)0, acc[nt], false, false);
    }
  }

  // C/D layout: VGPR r, lanes 0-15 -> (M=r, N=lane); lanes 16-31 -> (M=8+r).
  const int mrow = m_base + hi * 8;
  const bool do_tanh = (padd != nullptr);
#pragma unroll
  for (int nt = 0; nt < 4; ++nt) {
    int n = n_base + nt * 16 + lan15;
    float bb = bias[n] + (do_tanh ? padd[n] : 0.0f);
#pragma unroll
    for (int r = 0; r < 8; ++r) {
      float v = acc[nt][r] + bb;
      v = do_tanh ? tanhf(v) : fmaxf(v, 0.0f);
      size_t off = (size_t)(mrow + r) * DIM + n;
      if (out_f) out_f[off] = v;
      else       out_bf[off] = (__bf16)v;
    }
  }
}

// ---------------------------------------------------------------------------
extern "C" void kernel_launch(void* const* d_in, const int* in_sizes, int n_in,
                              void* d_out, int out_size, void* d_ws, size_t ws_size,
                              hipStream_t stream) {
  const float* x       = (const float*)d_in[0];
  const float* lin0_w  = (const float*)d_in[1];
  const float* lin0_b  = (const float*)d_in[2];
  const float* lin1_w  = (const float*)d_in[3];
  const float* lin1_b  = (const float*)d_in[4];
  const float* u1_w    = (const float*)d_in[5];
  const float* u1_cw   = (const float*)d_in[6];
  const float* u1_cb   = (const float*)d_in[7];
  const float* lin2_w  = (const float*)d_in[8];
  const float* lin2_b  = (const float*)d_in[9];
  const float* u2_w    = (const float*)d_in[10];
  const float* u2_cw   = (const float*)d_in[11];
  const float* u2_cb   = (const float*)d_in[12];
  const float* lino_w  = (const float*)d_in[13];
  const float* lino_b  = (const float*)d_in[14];

  // Workspace layout: actA(16MB) | actB(16MB) | 6 bf16 weights | p1 | p2
  char* ws = (char*)d_ws;
  const size_t actBytes = (size_t)BATCH * DIM * sizeof(__bf16);
  __bf16* actA = (__bf16*)ws;
  __bf16* actB = (__bf16*)(ws + actBytes);
  __bf16* Wbase = (__bf16*)(ws + 2 * actBytes);
  __bf16* W0 = Wbase + 0 * DD;
  __bf16* W1 = Wbase + 1 * DD;
  __bf16* W2 = Wbase + 2 * DD;
  __bf16* W3 = Wbase + 3 * DD;
  __bf16* W4 = Wbase + 4 * DD;
  __bf16* W5 = Wbase + 5 * DD;
  float* p1 = (float*)(ws + 2 * actBytes + 6 * DD * sizeof(__bf16));
  float* p2 = p1 + DIM;

  dim3 blk(256);

  // Weight conversion (tiny).  lin layers use W.T -> no transpose;
  // universal cw used untransposed -> transpose into Bt layout.
  prep_w_kernel<<<DD / 256, blk, 0, stream>>>(lin0_w, W0, 0);
  prep_w_kernel<<<DD / 256, blk, 0, stream>>>(lin1_w, W1, 0);
  prep_w_kernel<<<DD / 256, blk, 0, stream>>>(u1_cw,  W2, 1);
  prep_w_kernel<<<DD / 256, blk, 0, stream>>>(lin2_w, W3, 0);
  prep_w_kernel<<<DD / 256, blk, 0, stream>>>(u2_cw,  W4, 1);
  prep_w_kernel<<<DD / 256, blk, 0, stream>>>(lino_w, W5, 0);

  universal_p_kernel<<<DIM, DIM, 0, stream>>>(u1_w, p1);
  universal_p_kernel<<<DIM, DIM, 0, stream>>>(u2_w, p2);

  cvt_x_kernel<<<(BATCH * DIM) / (4 * 256), blk, 0, stream>>>(x, actA);

  dim3 gg(BATCH / 128, DIM / 64);   // (256, 4) blocks
  gemm_kernel<<<gg, blk, 0, stream>>>(actA, W0, lin0_b, nullptr, actB, nullptr); // relu
  gemm_kernel<<<gg, blk, 0, stream>>>(actB, W1, lin1_b, nullptr, actA, nullptr); // relu
  gemm_kernel<<<gg, blk, 0, stream>>>(actA, W2, u1_cb, p1,      actB, nullptr); // tanh(+p)
  gemm_kernel<<<gg, blk, 0, stream>>>(actB, W3, lin2_b, nullptr, actA, nullptr); // relu
  gemm_kernel<<<gg, blk, 0, stream>>>(actA, W4, u2_cb, p2,      actB, nullptr); // tanh(+p)
  gemm_kernel<<<gg, blk, 0, stream>>>(actB, W5, lino_b, nullptr, nullptr,
                                      (float*)d_out);                            // relu, f32
}
